// WindowAttention_16741782519872
// MI455X (gfx1250) — matrix operand order repaired and checked by reading, not compiled, not run on hardware
//
#include <hip/hip_runtime.h>
#include <hip/hip_bf16.h>

typedef __attribute__((ext_vector_type(16))) _Float16 v16h;
typedef __attribute__((ext_vector_type(8)))  float    v8f;
typedef __attribute__((ext_vector_type(4)))  unsigned int u32x4;

#define B_WIN 4096
#define NTOK  49
#define DIM   384
#define NH    12
#define HD    32
#define MROWS (B_WIN * NTOK)   // 200704

union FragU { v16h v; uint4 q[2]; _Float16 h[16]; };
union H8   { _Float16 h[8]; uint4 q; };
union U4H8 { u32x4 u; _Float16 h[8]; };

__device__ inline v8f wmma_f16(v16h a, v16h b, v8f c) {
  return __builtin_amdgcn_wmma_f32_16x16x32_f16(false, a, false, b, (short)0, c, false, false);
}

// Async global->LDS 16-byte copy (GLOBAL_LOAD_ASYNC_TO_LDS_B128, ASYNCcnt).
// GV addressing: vdst = LDS byte address, vaddr = 64-bit global address.
__device__ __forceinline__ void async_copy16(const void* g, void* lds) {
  uint32_t l = (uint32_t)(uintptr_t)lds;          // LDS aperture: low 32 bits = LDS offset
  uint64_t a = (uint64_t)(uintptr_t)g;
  asm volatile("global_load_async_to_lds_b128 %0, %1, off"
               :: "v"(l), "v"(a) : "memory");
}
__device__ __forceinline__ void async_wait0() {
  asm volatile("s_wait_asynccnt 0x0" ::: "memory");
}

// ---------------------------------------------------------------------------
// f32 -> f16 conversion (weights only), 8 elements / thread
// ---------------------------------------------------------------------------
__global__ __launch_bounds__(256) void f32_to_f16_kernel(const float* __restrict__ src,
                                                         _Float16* __restrict__ dst,
                                                         size_t n) {
  size_t stride = (size_t)gridDim.x * blockDim.x * 8;
  for (size_t i = ((size_t)blockIdx.x * blockDim.x + threadIdx.x) * 8; i < n; i += stride) {
    float4 f0 = *(const float4*)(src + i);
    float4 f1 = *(const float4*)(src + i + 4);
    H8 u;
    u.h[0] = (_Float16)f0.x; u.h[1] = (_Float16)f0.y;
    u.h[2] = (_Float16)f0.z; u.h[3] = (_Float16)f0.w;
    u.h[4] = (_Float16)f1.x; u.h[5] = (_Float16)f1.y;
    u.h[6] = (_Float16)f1.z; u.h[7] = (_Float16)f1.w;
    *(uint4*)(dst + i) = u.q;
  }
}

// ---------------------------------------------------------------------------
// Combined bias table: BM[w][h][64][64] = rpb[rel[i,j],h] + mask[w,i,j],
// padded with -1e30 for i>=49 or j>=49. 64*12*4096 floats (~12.6MB, L2 resident).
// ---------------------------------------------------------------------------
__global__ __launch_bounds__(256) void bias_table_kernel(const float* __restrict__ mask,
                                                         const float* __restrict__ rpb,
                                                         const int* __restrict__ rel,
                                                         float* __restrict__ bm) {
  int idx = blockIdx.x * 256 + threadIdx.x;          // 3,145,728 total
  int j = idx & 63;
  int i = (idx >> 6) & 63;
  int wh = idx >> 12;                                 // w*12 + h
  int w = wh / NH, h = wh - w * NH;
  float v = -1e30f;
  if (i < NTOK && j < NTOK)
    v = rpb[rel[i * NTOK + j] * NH + h] + mask[((size_t)w * NTOK + i) * NTOK + j];
  bm[idx] = v;
}

// ---------------------------------------------------------------------------
// GEMM: C[M,64] tile = A[m0:m0+64, 384] @ W[n0:n0+64, 384]^T
// MODE 0: A = fp32 x (convert in VGPRs), epilogue scatters f16 into qkv layout
// MODE 1: A = f16 attn_out (async-staged), epilogue writes fp32 out (+proj_b)
// 128 threads = 4 waves, each wave owns a 16x64 row strip.
// ---------------------------------------------------------------------------
template<int MODE>
__global__ __launch_bounds__(128) void gemm_kernel(const void* __restrict__ Aptr,
                                                   const _Float16* __restrict__ W,
                                                   const float* __restrict__ bias,
                                                   void* __restrict__ outp) {
  __shared__ __align__(16) _Float16 As[64 * 40];
  __shared__ __align__(16) _Float16 Ws[64 * 40];

  const int tid   = threadIdx.x;
  const int lane  = tid & 31;
  const int wave  = tid >> 5;
  const int m0    = blockIdx.x * 64;
  const int n0    = blockIdx.y * 64;
  const int strip = wave * 16;

  v8f acc[4];
#pragma unroll
  for (int t = 0; t < 4; ++t) acc[t] = (v8f){0.f, 0.f, 0.f, 0.f, 0.f, 0.f, 0.f, 0.f};

#pragma unroll 1
  for (int kt = 0; kt < 12; ++kt) {
    for (int c = tid; c < 256; c += 128) {
      int row = c >> 2, ch = c & 3;
      if (MODE == 0) {
        const float* A = (const float*)Aptr;
        const float* gp = A + (size_t)(m0 + row) * DIM + kt * 32 + ch * 8;
        float4 f0 = *(const float4*)gp;
        float4 f1 = *(const float4*)(gp + 4);
        H8 u;
        u.h[0] = (_Float16)f0.x; u.h[1] = (_Float16)f0.y;
        u.h[2] = (_Float16)f0.z; u.h[3] = (_Float16)f0.w;
        u.h[4] = (_Float16)f1.x; u.h[5] = (_Float16)f1.y;
        u.h[6] = (_Float16)f1.z; u.h[7] = (_Float16)f1.w;
        *(uint4*)(As + row * 40 + ch * 8) = u.q;
        if (kt + 1 < 12) __builtin_prefetch(gp + 32, 0, 0);
      } else {
        const _Float16* A = (const _Float16*)Aptr;
        async_copy16(A + (size_t)(m0 + row) * DIM + kt * 32 + ch * 8,
                     As + row * 40 + ch * 8);
      }
      async_copy16(W + (size_t)(n0 + row) * DIM + kt * 32 + ch * 8,
                   Ws + row * 40 + ch * 8);
    }
    async_wait0();
    __syncthreads();

    // A fragment + all 4 B fragments live together -> clause the ds_loads,
    // then 4 independent WMMAs back-to-back (no per-WMMA dscnt stall).
    FragU a, bf0, bf1, bf2, bf3;
    {
      const _Float16* ap = As + (strip + (lane & 15)) * 40 + ((lane >> 4) << 4);
      a.q[0] = *(const uint4*)ap;         // K = kb .. kb+7
      a.q[1] = *(const uint4*)(ap + 8);   // K = kb+8 .. kb+15
      const _Float16* bp = Ws + (lane & 15) * 40 + ((lane >> 4) << 4);
      bf0.q[0] = *(const uint4*)bp;              bf0.q[1] = *(const uint4*)(bp + 8);
      bf1.q[0] = *(const uint4*)(bp + 16 * 40);  bf1.q[1] = *(const uint4*)(bp + 16 * 40 + 8);
      bf2.q[0] = *(const uint4*)(bp + 32 * 40);  bf2.q[1] = *(const uint4*)(bp + 32 * 40 + 8);
      bf3.q[0] = *(const uint4*)(bp + 48 * 40);  bf3.q[1] = *(const uint4*)(bp + 48 * 40 + 8);
    }
    acc[0] = wmma_f16(a.v, bf0.v, acc[0]);
    acc[1] = wmma_f16(a.v, bf1.v, acc[1]);
    acc[2] = wmma_f16(a.v, bf2.v, acc[2]);
    acc[3] = wmma_f16(a.v, bf3.v, acc[3]);
    __syncthreads();
  }

  const int eOff = (lane >> 4) << 3;  // C layout: M = element + 8*(lane>=16)
  const int colL = lane & 15;
#pragma unroll
  for (int tn = 0; tn < 4; ++tn) {
    int n = n0 + tn * 16 + colL;
    float bv = bias[n];
    if (MODE == 0) {
      _Float16* qkv16 = (_Float16*)outp;
      int which = n / DIM;
      int rem = n - which * DIM;
      int h = rem >> 5, d = rem & 31;
#pragma unroll
      for (int e = 0; e < 8; ++e) {
        int m = m0 + strip + eOff + e;
        int b = m / NTOK, t = m - b * NTOK;
        size_t dst = (((size_t)which * B_WIN + b) * NH + h) * (NTOK * HD) + (size_t)t * HD + d;
        __builtin_nontemporal_store((_Float16)(acc[tn][e] + bv), qkv16 + dst);
      }
    } else {
      float* out = (float*)outp;
#pragma unroll
      for (int e = 0; e < 8; ++e) {
        int m = m0 + strip + eOff + e;
        __builtin_nontemporal_store(acc[tn][e] + bv, out + (size_t)m * DIM + n);
      }
    }
  }
}

// ---------------------------------------------------------------------------
// Fused attention: one block (128 thr = 4 waves) per (window b, head h).
// S = scale*q@k^T + BM -> softmax -> O = P@V, all in LDS/WMMA.
// Tokens padded 49 -> 64 (4 M-strips of 16, one per wave).
// ---------------------------------------------------------------------------
__global__ __launch_bounds__(128) void attn_kernel(const _Float16* __restrict__ qkv,
                                                   const float* __restrict__ bm,
                                                   _Float16* __restrict__ attn_out) {
  const int bh = blockIdx.x;
  const int b = bh / NH, h = bh - b * NH;
  const int tid = threadIdx.x, lane = tid & 31, wave = tid >> 5;

  __shared__ __align__(16) _Float16 qs[64 * 40];  // q rows (token, hd)
  __shared__ __align__(16) _Float16 ks[64 * 40];  // k rows (token, hd)
  __shared__ __align__(16) _Float16 vt[32 * 72];  // v transposed (hd, token)
  __shared__ __align__(16) _Float16 ps[64 * 72];  // softmax probs (token, token)

  // zero only what can NaN-poison stored outputs: k pad rows, v pad columns
  for (int i = tid; i < 64 * 40; i += 128) ks[i] = (_Float16)0.f;
  for (int i = tid; i < 32 * 72; i += 128) vt[i] = (_Float16)0.f;
  __syncthreads();

  const size_t head = ((size_t)b * NH + h) * (NTOK * HD);
  const _Float16* qg = qkv + head;
  const _Float16* kg = qkv + (size_t)B_WIN * NH * NTOK * HD + head;
  const _Float16* vg = qkv + 2 * (size_t)B_WIN * NH * NTOK * HD + head;

  for (int c = tid; c < NTOK * 4; c += 128) {      // 16B chunks of 49x32 rows
    int t = c >> 2, ch = c & 3;
    async_copy16(qg + t * 32 + ch * 8, qs + t * 40 + ch * 8);
    async_copy16(kg + t * 32 + ch * 8, ks + t * 40 + ch * 8);
    U4H8 u;
    u.u = __builtin_nontemporal_load((const u32x4*)(vg + t * 32 + ch * 8));
#pragma unroll
    for (int d = 0; d < 8; ++d) vt[(ch * 8 + d) * 72 + t] = u.h[d];
  }
  async_wait0();
  __syncthreads();

  const int strip = wave * 16;
  const int eOff = (lane >> 4) << 3;
  const int j0 = lane & 15;

  // --- S = q @ k^T (K = 32): load all fragments, then 4 WMMAs back-to-back ---
  v8f acc[4];
#pragma unroll
  for (int t = 0; t < 4; ++t) acc[t] = (v8f){0.f, 0.f, 0.f, 0.f, 0.f, 0.f, 0.f, 0.f};
  {
    FragU a, k0, k1, k2, k3;
    const _Float16* ap = qs + (strip + (lane & 15)) * 40 + ((lane >> 4) << 4);
    a.q[0] = *(const uint4*)ap;
    a.q[1] = *(const uint4*)(ap + 8);
    const _Float16* bp = ks + (lane & 15) * 40 + ((lane >> 4) << 4);
    k0.q[0] = *(const uint4*)bp;              k0.q[1] = *(const uint4*)(bp + 8);
    k1.q[0] = *(const uint4*)(bp + 16 * 40);  k1.q[1] = *(const uint4*)(bp + 16 * 40 + 8);
    k2.q[0] = *(const uint4*)(bp + 32 * 40);  k2.q[1] = *(const uint4*)(bp + 32 * 40 + 8);
    k3.q[0] = *(const uint4*)(bp + 48 * 40);  k3.q[1] = *(const uint4*)(bp + 48 * 40 + 8);
    acc[0] = wmma_f16(a.v, k0.v, acc[0]);
    acc[1] = wmma_f16(a.v, k1.v, acc[1]);
    acc[2] = wmma_f16(a.v, k2.v, acc[2]);
    acc[3] = wmma_f16(a.v, k3.v, acc[3]);
  }

  // --- scale + combined bias/mask table (branch-free, coalesced) ---
  const float scale = 0.1767766952966369f;  // 32^-0.5
  const float* bmp = bm + (((size_t)(b & 63) * NH + h) << 12);  // [64][64] tile
#pragma unroll
  for (int e = 0; e < 8; ++e) {
    int i = strip + eOff + e;
#pragma unroll
    for (int tn = 0; tn < 4; ++tn)
      acc[tn][e] = acc[tn][e] * scale + bmp[i * 64 + tn * 16 + j0];
  }

  // --- softmax per row (row = element, cols across 4 tiles x 16 lanes) ---
#pragma unroll
  for (int e = 0; e < 8; ++e) {
    float mx = fmaxf(fmaxf(acc[0][e], acc[1][e]), fmaxf(acc[2][e], acc[3][e]));
#pragma unroll
    for (int s = 1; s < 16; s <<= 1) mx = fmaxf(mx, __shfl_xor(mx, s, 32));
    float p[4], sum = 0.f;
#pragma unroll
    for (int tn = 0; tn < 4; ++tn) { p[tn] = __expf(acc[tn][e] - mx); sum += p[tn]; }
#pragma unroll
    for (int s = 1; s < 16; s <<= 1) sum += __shfl_xor(sum, s, 32);
    float inv = 1.f / (sum + 1e-20f);
    int i = strip + eOff + e;
#pragma unroll
    for (int tn = 0; tn < 4; ++tn)
      ps[i * 72 + tn * 16 + j0] = (_Float16)(p[tn] * inv);
  }
  __syncthreads();

  // --- O = P @ V (K = 64): load all fragments, then 4 WMMAs back-to-back ---
  v8f oacc[2];
  oacc[0] = (v8f){0.f, 0.f, 0.f, 0.f, 0.f, 0.f, 0.f, 0.f};
  oacc[1] = oacc[0];
  {
    FragU pa0, pa1, vb00, vb01, vb10, vb11;
    const _Float16* pp = ps + (strip + (lane & 15)) * 72 + ((lane >> 4) << 4);
    pa0.q[0] = *(const uint4*)pp;        pa0.q[1] = *(const uint4*)(pp + 8);
    pa1.q[0] = *(const uint4*)(pp + 32); pa1.q[1] = *(const uint4*)(pp + 40);
    const _Float16* vp = vt + (lane & 15) * 72 + ((lane >> 4) << 4);
    vb00.q[0] = *(const uint4*)vp;                  vb00.q[1] = *(const uint4*)(vp + 8);
    vb01.q[0] = *(const uint4*)(vp + 32);           vb01.q[1] = *(const uint4*)(vp + 40);
    vb10.q[0] = *(const uint4*)(vp + 16 * 72);      vb10.q[1] = *(const uint4*)(vp + 16 * 72 + 8);
    vb11.q[0] = *(const uint4*)(vp + 16 * 72 + 32); vb11.q[1] = *(const uint4*)(vp + 16 * 72 + 40);
    oacc[0] = wmma_f16(pa0.v, vb00.v, oacc[0]);
    oacc[1] = wmma_f16(pa0.v, vb10.v, oacc[1]);
    oacc[0] = wmma_f16(pa1.v, vb01.v, oacc[0]);
    oacc[1] = wmma_f16(pa1.v, vb11.v, oacc[1]);
  }

#pragma unroll
  for (int tn = 0; tn < 2; ++tn) {
    int d = tn * 16 + (lane & 15);
#pragma unroll
    for (int e = 0; e < 8; ++e) {
      int i = strip + eOff + e;
      if (i < NTOK)
        __builtin_nontemporal_store((_Float16)oacc[tn][e],
                                    attn_out + ((size_t)b * NTOK + i) * DIM + h * HD + d);
    }
  }
}

// ---------------------------------------------------------------------------
extern "C" void kernel_launch(void* const* d_in, const int* in_sizes, int n_in,
                              void* d_out, int out_size, void* d_ws, size_t ws_size,
                              hipStream_t stream) {
  const float* x      = (const float*)d_in[0];
  const float* mask   = (const float*)d_in[1];
  const float* qkv_w  = (const float*)d_in[2];
  const float* qkv_b  = (const float*)d_in[3];
  const float* proj_w = (const float*)d_in[4];
  const float* proj_b = (const float*)d_in[5];
  const float* rpb    = (const float*)d_in[6];
  const int*   rel    = (const int*)d_in[7];
  float* out = (float*)d_out;

  char* ws = (char*)d_ws;
  const size_t ATTN_BYTES = (size_t)MROWS * DIM * 2;               // 154.1 MB f16
  const size_t QKV_BYTES  = 3ull * B_WIN * NH * NTOK * HD * 2;     // 462.4 MB f16
  const size_t W16_BYTES  = (size_t)3 * DIM * DIM * 2;
  const size_t PW16_BYTES = (size_t)DIM * DIM * 2;
  _Float16* attn16 = (_Float16*)ws;
  _Float16* qkv16  = (_Float16*)(ws + ATTN_BYTES);
  _Float16* qw16   = (_Float16*)(ws + ATTN_BYTES + QKV_BYTES);
  _Float16* pw16   = (_Float16*)(ws + ATTN_BYTES + QKV_BYTES + W16_BYTES);
  float*    bmtab  = (float*)(ws + ATTN_BYTES + QKV_BYTES + W16_BYTES + PW16_BYTES);

  // weight conversions (tiny) + combined bias/mask table
  f32_to_f16_kernel<<<216, 256, 0, stream>>>(qkv_w, qw16, (size_t)3 * DIM * DIM);
  f32_to_f16_kernel<<<72, 256, 0, stream>>>(proj_w, pw16, (size_t)DIM * DIM);
  bias_table_kernel<<<(64 * NH * 4096) / 256, 256, 0, stream>>>(mask, rpb, rel, bmtab);

  // qkv projection: [200704,384] @ [1152,384]^T
  gemm_kernel<0><<<dim3(MROWS / 64, (3 * DIM) / 64), 128, 0, stream>>>(x, qw16, qkv_b, qkv16);

  // fused window attention per (window, head)
  attn_kernel<<<B_WIN * NH, 128, 0, stream>>>(qkv16, bmtab, attn16);

  // output projection: [200704,384] @ [384,384]^T
  gemm_kernel<1><<<dim3(MROWS / 64, DIM / 64), 128, 0, stream>>>(attn16, pw16, proj_b, out);
}
